// RV2BEVFrustumAttn_37280316129321
// MI455X (gfx1250) — matrix operand order repaired and checked by reading, not compiled, not run on hardware
//
#include <hip/hip_runtime.h>
#include <hip/hip_bf16.h>
#include <math.h>

// ---------------- static problem config ----------------
constexpr int kB    = 2;
constexpr int kHRV  = 32;
constexpr int kWRV  = 1024;
constexpr int kNQ   = kHRV * kWRV;        // 32768 per batch
constexpr int kRows = kB * kNQ;           // 65536 query rows
constexpr int kD    = 128;
constexpr int kCB   = 256;
constexpr int kHB   = 200;
constexpr int kWB   = 200;
constexpr int kNV   = kHB * kWB;          // 40000 per batch
constexpr int kVRows = kB * kNV;          // 80000 value rows
constexpr int kHeads = 8;
constexpr int kPts   = 6;
constexpr int kDH    = 16;
constexpr int kPad   = 16;                // LDS column pad (halves) = 32B

typedef __attribute__((ext_vector_type(16))) __bf16 v16bf;
typedef __attribute__((ext_vector_type(8)))  __bf16 v8bf;
typedef __attribute__((ext_vector_type(8)))  float  v8f;
typedef unsigned short u16;

// ---------------- scalar helpers ----------------
__device__ __forceinline__ u16 f2bf(float f) {
  unsigned int u = __float_as_uint(f);
  u += 0x7fffu + ((u >> 16) & 1u);          // round-to-nearest-even
  return (u16)(u >> 16);
}
__device__ __forceinline__ float bf2f(u16 h) {
  return __uint_as_float(((unsigned int)h) << 16);
}
__device__ __forceinline__ float gelu_exact(float x) {
  return 0.5f * x * (1.0f + erff(x * 0.70710678118654752440f));
}

union AFrag { v16bf v; v8bf p[2]; };

// cooperative global -> LDS copy of Bt[N,Kp] into Bs with padded stride Kps
__device__ __forceinline__ void fill_lds_b(u16* Bs, const u16* __restrict__ Bt,
                                           int N, int Kp, int Kps) {
  const int kc = Kp >> 3;                  // uint4 chunks per column
  const int chunks = N * kc;
  for (int j = threadIdx.x; j < chunks; j += 256) {
    const int n = j / kc, r = j - n * kc;
    *(uint4*)(Bs + (size_t)n * Kps + r * 8) =
        *(const uint4*)(Bt + (size_t)n * Kp + r * 8);
  }
  __syncthreads();
}

// epilogue flags
enum { EPI_BIAS = 1, EPI_GELU = 2, EPI_WRF = 4, EPI_WRB = 8 };

// ---------------- generic bf16 WMMA GEMM ----------------
// C[M,N] = A[M,KP](bf16) * Bt[N,KP](bf16, K-contiguous) (+bias)(+gelu)
// B staged in LDS (padded stride), A register ping-pong one k-step ahead,
// per-step B fragments preloaded into a register array so the WMMA chain
// pipelines on descending dscnt instead of draining per instruction.
// 256 threads = 8 waves; wave owns a 16-row strip, NTILES 16-col tiles.
// Requires gridDim.x * 128 == M (all M here are multiples of 128).
template<int NTILES, int KP, int EPI>
__global__ __launch_bounds__(256)
void k_gemm_bf16(const u16* __restrict__ A, const u16* __restrict__ Bt,
                 const float* __restrict__ bias,
                 float* __restrict__ Cf, u16* __restrict__ Cb, int ldc) {
  extern __shared__ u16 Bs[];
  constexpr int Kps = KP + kPad;
  constexpr int NSTEP = KP / 32;
  fill_lds_b(Bs, Bt, NTILES * 16, KP, Kps);

  const int lane = threadIdx.x & 31;
  const int wave = threadIdx.x >> 5;
  const int row0 = blockIdx.x * 128 + wave * 16;
  const int nsel = lane & 15;
  const int hi   = lane >> 4;   // 0: lanes 0-15, 1: lanes 16-31

  v8f acc[NTILES];
#pragma unroll
  for (int t = 0; t < NTILES; ++t) acc[t] = {};

  const u16* aptr  = A + (size_t)(row0 + nsel) * KP + hi * 16;
  const u16* bbase = Bs + (size_t)nsel * Kps + hi * 16;

  AFrag af[2];
  af[0].p[0] = *(const v8bf*)(aptr);
  af[0].p[1] = *(const v8bf*)(aptr + 8);

#pragma unroll
  for (int s = 0; s < NSTEP; ++s) {
    const int cur = s & 1;
    if (s + 1 < NSTEP) {                       // prefetch next A fragment
      const u16* np = aptr + (s + 1) * 32;
      af[cur ^ 1].p[0] = *(const v8bf*)(np);
      af[cur ^ 1].p[1] = *(const v8bf*)(np + 8);
    }
    v16bf bf[NTILES];                          // preload all B frags (const offs)
#pragma unroll
    for (int t = 0; t < NTILES; ++t)
      bf[t] = *(const v16bf*)(bbase + t * 16 * Kps + s * 32);
#pragma unroll
    for (int t = 0; t < NTILES; ++t)
      acc[t] = __builtin_amdgcn_wmma_f32_16x16x32_bf16(
          false, af[cur].v, false, bf[t], (short)0, acc[t], false, false);
  }

#pragma unroll
  for (int t = 0; t < NTILES; ++t) {
#pragma unroll
    for (int r = 0; r < 8; ++r) {
      const int m = row0 + r + hi * 8;
      const int n = t * 16 + nsel;
      float v = acc[t][r];
      if constexpr (EPI & EPI_BIAS) v += bias[n];
      if constexpr (EPI & EPI_GELU) v = gelu_exact(v);
      const size_t idx = (size_t)m * ldc + n;
      if constexpr (EPI & EPI_WRF) Cf[idx] = v;
      if constexpr (EPI & EPI_WRB) Cb[idx] = f2bf(v);
    }
  }
}

// ---------------- 3x3 circular conv as 9 tap-shifted WMMA GEMMs ----------------
// Hg: [kRows,128] bf16 (post GN+GELU), Wt: [64, 9*128] bf16 (Wt[n][tap*128+i])
// Weights staged in LDS; A fragment ping-pong, B frags preloaded per chunk,
// fully unrolled 36-chunk K loop.
__global__ __launch_bounds__(256)
void k_conv3x3_wmma(const u16* __restrict__ Hg, const u16* __restrict__ Wt,
                    float* __restrict__ H2) {
  extern __shared__ u16 Bs[];
  constexpr int Kp = 1152, Kps = Kp + kPad;
  fill_lds_b(Bs, Wt, 64, Kp, Kps);

  const int lane = threadIdx.x & 31;
  const int wave = threadIdx.x >> 5;
  const int row0 = blockIdx.x * 128 + wave * 16;
  const int nsel = lane & 15;
  const int hi   = lane >> 4;
  const int arow = row0 + nsel;
  const int b = arow >> 15;
  const int h = (arow >> 10) & 31;
  const int w = arow & 1023;

  // precompute circularly-shifted source rows for the 9 taps
  int rows[9];
#pragma unroll
  for (int dy = -1; dy <= 1; ++dy)
#pragma unroll
    for (int dx = -1; dx <= 1; ++dx) {
      const int hh = (h + dy + kHRV) & (kHRV - 1);
      const int ww = (w + dx + kWRV) & (kWRV - 1);
      rows[(dy + 1) * 3 + (dx + 1)] = (b << 15) + (hh << 10) + ww;
    }

  v8f acc[4];
#pragma unroll
  for (int t = 0; t < 4; ++t) acc[t] = {};

  const u16* bbase = Bs + (size_t)nsel * Kps + hi * 16;

  auto aChunk = [&](int chunk) -> const u16* {
    // chunk = tap*4 + (k0/32), per-tap K = 128
    return Hg + (size_t)rows[chunk >> 2] * 128 + ((chunk & 3) << 5) + hi * 16;
  };

  AFrag af[2];
  af[0].p[0] = *(const v8bf*)(aChunk(0));
  af[0].p[1] = *(const v8bf*)(aChunk(0) + 8);
#pragma unroll
  for (int chunk = 0; chunk < 36; ++chunk) {
    const int cur = chunk & 1;
    if (chunk + 1 < 36) {
      const u16* np = aChunk(chunk + 1);
      af[cur ^ 1].p[0] = *(const v8bf*)(np);
      af[cur ^ 1].p[1] = *(const v8bf*)(np + 8);
    }
    v16bf bf[4];
#pragma unroll
    for (int t = 0; t < 4; ++t)
      bf[t] = *(const v16bf*)(bbase + t * 16 * Kps + chunk * 32);
#pragma unroll
    for (int t = 0; t < 4; ++t)
      acc[t] = __builtin_amdgcn_wmma_f32_16x16x32_bf16(
          false, af[cur].v, false, bf[t], (short)0, acc[t], false, false);
  }
#pragma unroll
  for (int t = 0; t < 4; ++t)
#pragma unroll
    for (int r = 0; r < 8; ++r)
      H2[(size_t)(row0 + r + hi * 8) * 64 + t * 16 + nsel] = acc[t][r];
}

// ---------------- GroupNorm (+GELU, bf16 out); one block per (batch,group) ----
__global__ __launch_bounds__(256)
void k_gn_gelu(const float* __restrict__ x, const float* __restrict__ gamma,
               const float* __restrict__ beta, u16* __restrict__ out,
               int C, int Cg) {
  __shared__ float s1[256], s2[256];
  const int G = C / Cg;
  const int b = blockIdx.x / G;
  const int g = blockIdx.x % G;
  const int tid = threadIdx.x;
  const int n = kNQ * Cg;
  float ps = 0.f, pq = 0.f;
  for (int j = tid; j < n; j += 256) {
    const int p = j / Cg, c = g * Cg + (j % Cg);
    const float v = x[(size_t)(b * kNQ + p) * C + c];
    ps += v; pq += v * v;
  }
  s1[tid] = ps; s2[tid] = pq;
  __syncthreads();
  for (int s = 128; s > 0; s >>= 1) {
    if (tid < s) { s1[tid] += s1[tid + s]; s2[tid] += s2[tid + s]; }
    __syncthreads();
  }
  const float mean = s1[0] / (float)n;
  const float var  = s2[0] / (float)n - mean * mean;
  const float inv  = rsqrtf(var + 1e-5f);
  for (int j = tid; j < n; j += 256) {
    const int p = j / Cg, c = g * Cg + (j % Cg);
    const size_t idx = (size_t)(b * kNQ + p) * C + c;
    const float v = (x[idx] - mean) * inv * gamma[c] + beta[c];
    out[idx] = f2bf(gelu_exact(v));
  }
}

// ---------------- range head tail: rh = H2g@w3+b3 -> ref pts + sigma feats ----
__global__ void k_range_head(const u16* __restrict__ H2g,
                             const float* __restrict__ w3, const float* __restrict__ b3,
                             float* __restrict__ ref, u16* __restrict__ Qcat) {
  const int row = blockIdx.x * blockDim.x + threadIdx.x;
  if (row >= kRows) return;
  float r0 = b3[0], r1 = b3[1];
  const u16* hp = H2g + (size_t)row * 64;
  for (int k = 0; k < 64; ++k) {
    const float v = bf2f(hp[k]);
    r0 += v * w3[2 * k];
    r1 += v * w3[2 * k + 1];
  }
  const float mu = fminf(fmaxf(r0, 0.f), 55.f);
  const float ls = fminf(fmaxf(r1, -5.f), 3.f);
  const float sg = expf(ls);
  const int w = row & (kWRV - 1);
  const float az = -3.14159265358979323846f +
                   6.28318530717958647692f * (float)w / (float)kWRV;
  const float xm = mu * cosf(az), ym = mu * sinf(az);
  ref[row * 2 + 0] = fminf(fmaxf((xm + 50.f) * 0.01f, 0.f), 1.f);
  ref[row * 2 + 1] = fminf(fmaxf((ym + 50.f) * 0.01f, 0.f), 1.f);
  u16* qp = Qcat + (size_t)row * 160;
  qp[128] = f2bf(ls);
  qp[129] = f2bf(1.0f / (sg + 1e-6f));
  for (int c = 130; c < 160; ++c) qp[c] = 0;   // zero K-pad
}

// ---------------- softmax over points per (row,head), in-place ----------------
__global__ void k_softmax_p(float* __restrict__ a) {
  const int t = blockIdx.x * blockDim.x + threadIdx.x;
  if (t >= kRows * kHeads) return;
  float* p = a + (size_t)(t / kHeads) * (kHeads * kPts) + (t % kHeads) * kPts;
  float m = p[0];
  for (int i = 1; i < kPts; ++i) m = fmaxf(m, p[i]);
  float e[kPts], s = 0.f;
  for (int i = 0; i < kPts; ++i) { e[i] = expf(p[i] - m); s += e[i]; }
  const float inv = 1.f / s;
  for (int i = 0; i < kPts; ++i) p[i] = e[i] * inv;
}

// ---------------- deformable bilinear sampling; thread per (row,head,dh) ------
__global__ __launch_bounds__(256)
void k_msda(const float* __restrict__ val, const float* __restrict__ ref,
            const float* __restrict__ offs, const float* __restrict__ attw,
            u16* __restrict__ outb) {
  const int tid = blockIdx.x * 256 + threadIdx.x;
  if (tid >= kRows * kD) return;
  const int d    = tid & 15;
  const int head = (tid >> 4) & 7;
  const int row  = tid >> 7;
  const int b    = row >> 15;
  const float* vb = val + (size_t)b * kNV * kD;
  const float rx = ref[row * 2], ry = ref[row * 2 + 1];
  float acc = 0.f;
  for (int p = 0; p < kPts; ++p) {
    const float ox = offs[(size_t)row * 96 + (head * kPts + p) * 2 + 0];
    const float oy = offs[(size_t)row * 96 + (head * kPts + p) * 2 + 1];
    const float px = (rx + ox * (1.f / kWB)) * kWB - 0.5f;
    const float py = (ry + oy * (1.f / kHB)) * kHB - 0.5f;
    const float fx = floorf(px), fy = floorf(py);
    const int x0 = (int)fx, y0 = (int)fy;
    const float wx = px - fx, wy = py - fy;
    float s = 0.f;
#pragma unroll
    for (int cy = 0; cy < 2; ++cy) {
      const int yi = y0 + cy;
      if (yi < 0 || yi >= kHB) continue;
      const float wyv = cy ? wy : (1.f - wy);
#pragma unroll
      for (int cx = 0; cx < 2; ++cx) {
        const int xi = x0 + cx;
        if (xi < 0 || xi >= kWB) continue;
        s += wyv * (cx ? wx : (1.f - wx)) *
             vb[(size_t)(yi * kWB + xi) * kD + head * kDH + d];
      }
    }
    acc += s * attw[(size_t)row * 48 + head * kPts + p];
  }
  outb[tid] = f2bf(acc);
}

// ---------------- weight transpose+pad+convert: W[K,N]f32 -> Wt[N,Kp]bf16 -----
__global__ void k_wt_bf16(const float* __restrict__ W, u16* __restrict__ Wt,
                          int K, int N, int Kp) {
  const int tid = blockIdx.x * blockDim.x + threadIdx.x;
  if (tid >= N * Kp) return;
  const int n = tid / Kp, k = tid % Kp;
  Wt[tid] = (k < K) ? f2bf(W[(size_t)k * N + n]) : (u16)0;
}

// ---------------- build Xa[row,64] and Xc[row,96] (x_rv ++ sin/cos/0, pad) ----
__global__ void k_prep_x(const float* __restrict__ x_rv,
                         u16* __restrict__ Xa, u16* __restrict__ Xc) {
  const int tid = blockIdx.x * blockDim.x + threadIdx.x;
  if (tid >= kRows * 96) return;
  const int c = tid % 96, row = tid / 96;
  float v = 0.f;
  if (c < 64) {
    v = x_rv[(size_t)row * 64 + c];
    Xa[(size_t)row * 64 + c] = f2bf(v);
  } else if (c == 64 || c == 65) {
    const int w = row & (kWRV - 1);
    const float az = -3.14159265358979323846f +
                     6.28318530717958647692f * (float)w / (float)kWRV;
    v = (c == 64) ? sinf(az) : cosf(az);
  }
  Xc[tid] = f2bf(v);
}

// ---------------- bev NCHW -> Va[(b*NV+p), 256] bf16, LDS-tiled transpose -----
__global__ __launch_bounds__(256)
void k_prep_va(const float* __restrict__ bev, u16* __restrict__ Va) {
  __shared__ float tile[32][33];
  const int b  = blockIdx.z;
  const int c0 = blockIdx.y * 32;
  const int p0 = blockIdx.x * 32;
  for (int i = threadIdx.y; i < 32; i += 8)
    tile[i][threadIdx.x] = bev[(size_t)(b * kCB + c0 + i) * kNV + p0 + threadIdx.x];
  __syncthreads();
  for (int i = threadIdx.y; i < 32; i += 8)
    Va[(size_t)(b * kNV + p0 + i) * kCB + c0 + threadIdx.x] =
        f2bf(tile[threadIdx.x][i]);
}

// =============================== host side ===================================
extern "C" void kernel_launch(void* const* d_in, const int* in_sizes, int n_in,
                              void* d_out, int out_size, void* d_ws, size_t ws_size,
                              hipStream_t stream) {
  const float* x_rv  = (const float*)d_in[0];
  const float* bev   = (const float*)d_in[1];
  const float* pq_w  = (const float*)d_in[2];  const float* pq_b  = (const float*)d_in[3];
  const float* pv_w  = (const float*)d_in[4];  const float* pv_b  = (const float*)d_in[5];
  const float* po_w  = (const float*)d_in[6];  const float* po_b  = (const float*)d_in[7];
  const float* qs_w1 = (const float*)d_in[8];  const float* qs_b1 = (const float*)d_in[9];
  const float* qs_w2 = (const float*)d_in[10]; const float* qs_b2 = (const float*)d_in[11];
  const float* rh_w1 = (const float*)d_in[12]; const float* rh_b1 = (const float*)d_in[13];
  const float* rh_g1 = (const float*)d_in[14]; const float* rh_be1= (const float*)d_in[15];
  const float* rh_w2 = (const float*)d_in[16];
  const float* rh_g2 = (const float*)d_in[17]; const float* rh_be2= (const float*)d_in[18];
  const float* rh_w3 = (const float*)d_in[19]; const float* rh_b3 = (const float*)d_in[20];
  const float* so_w  = (const float*)d_in[21]; const float* so_b  = (const float*)d_in[22];
  const float* aw_w  = (const float*)d_in[23]; const float* aw_b  = (const float*)d_in[24];
  const float* vp_w  = (const float*)d_in[25]; const float* vp_b  = (const float*)d_in[26];
  const float* op_w  = (const float*)d_in[27]; const float* op_b  = (const float*)d_in[28];
  (void)in_sizes; (void)n_in; (void)out_size; (void)ws_size;

  char* ws = (char*)d_ws;
  size_t off = 0;
  auto alloc = [&](size_t bytes) -> void* {
    void* p = ws + off;
    off += (bytes + 255) & ~(size_t)255;
    return p;
  };

  // bf16 transposed (N,Kp) weights
  u16* pq_wT  = (u16*)alloc((size_t)128 * 64 * 2);
  u16* rh_w1T = (u16*)alloc((size_t)128 * 96 * 2);
  u16* rh_w2T = (u16*)alloc((size_t)64 * 1152 * 2);
  u16* qs_w1T = (u16*)alloc((size_t)128 * 160 * 2);
  u16* qs_w2T = (u16*)alloc((size_t)128 * 128 * 2);
  u16* pv_wT  = (u16*)alloc((size_t)128 * 256 * 2);
  u16* vp_wT  = (u16*)alloc((size_t)128 * 128 * 2);
  u16* so_wT  = (u16*)alloc((size_t)96 * 128 * 2);
  u16* aw_wT  = (u16*)alloc((size_t)48 * 128 * 2);
  u16* op_wT  = (u16*)alloc((size_t)128 * 128 * 2);
  u16* po_wT  = (u16*)alloc((size_t)128 * 128 * 2);
  // activations
  u16*   Xa    = (u16*)  alloc((size_t)kRows * 64 * 2);
  u16*   Xc    = (u16*)  alloc((size_t)kRows * 96 * 2);
  u16*   Qcat  = (u16*)  alloc((size_t)kRows * 160 * 2);   // [Q0 | ls | 1/sig | pad]
  float* H1    = (float*)alloc((size_t)kRows * 128 * 4);
  u16*   H1g   = (u16*)  alloc((size_t)kRows * 128 * 2);
  float* H2    = (float*)alloc((size_t)kRows * 64 * 4);
  u16*   H2g   = (u16*)  alloc((size_t)kRows * 64 * 2);
  float* refb  = (float*)alloc((size_t)kRows * 2 * 4);
  u16*   Q1g   = (u16*)  alloc((size_t)kRows * 128 * 2);
  u16*   Qry   = (u16*)  alloc((size_t)kRows * 128 * 2);
  u16*   Va    = (u16*)  alloc((size_t)kVRows * 256 * 2);
  u16*   Vg    = (u16*)  alloc((size_t)kVRows * 128 * 2);
  float* value = (float*)alloc((size_t)kVRows * 128 * 4);
  float* offsb = (float*)alloc((size_t)kRows * 96 * 4);
  float* attl  = (float*)alloc((size_t)kRows * 48 * 4);
  u16*   outb  = (u16*)  alloc((size_t)kRows * 128 * 2);
  u16*   o2b   = (u16*)  alloc((size_t)kRows * 128 * 2);

  auto wt = [&](const float* W, u16* Wt, int K, int N, int Kp) {
    const int tot = N * Kp;
    k_wt_bf16<<<(tot + 255) / 256, 256, 0, stream>>>(W, Wt, K, N, Kp);
  };
  wt(pq_w,  pq_wT,  64,   128, 64);
  wt(rh_w1, rh_w1T, 67,   128, 96);
  wt(rh_w2, rh_w2T, 1152, 64,  1152);
  wt(qs_w1, qs_w1T, 130,  128, 160);
  wt(qs_w2, qs_w2T, 128,  128, 128);
  wt(pv_w,  pv_wT,  256,  128, 256);
  wt(vp_w,  vp_wT,  128,  128, 128);
  wt(so_w,  so_wT,  128,  96,  128);
  wt(aw_w,  aw_wT,  128,  48,  128);
  wt(op_w,  op_wT,  128,  128, 128);
  wt(po_w,  po_wT,  128,  128, 128);

  k_prep_x<<<(kRows * 96 + 255) / 256, 256, 0, stream>>>(x_rv, Xa, Xc);
  k_prep_va<<<dim3(kNV / 32, kCB / 32, kB), dim3(32, 8), 0, stream>>>(bev, Va);

  const dim3 gQ(kRows / 128), gV(kVRows / 128), blk(256);
  auto sh = [](int nt, int kp) { return (size_t)nt * 16 * (kp + kPad) * 2; };

  // Q0 -> bf16 into Qcat cols 0..127 (ldc=160)
  k_gemm_bf16<8, 64, EPI_BIAS | EPI_WRB><<<gQ, blk, sh(8, 64), stream>>>(
      Xa, pq_wT, pq_b, nullptr, Qcat, 160);
  // range head GEMM1 -> f32 H1
  k_gemm_bf16<8, 96, EPI_BIAS | EPI_WRF><<<gQ, blk, sh(8, 96), stream>>>(
      Xc, rh_w1T, rh_b1, H1, nullptr, 128);
  k_gn_gelu<<<kB * 8, blk, 0, stream>>>(H1, rh_g1, rh_be1, H1g, 128, 16);
  k_conv3x3_wmma<<<gQ, blk, (size_t)64 * (1152 + kPad) * 2, stream>>>(H1g, rh_w2T, H2);
  k_gn_gelu<<<kB * 8, blk, 0, stream>>>(H2, rh_g2, rh_be2, H2g, 64, 8);
  k_range_head<<<kRows / 256, blk, 0, stream>>>(H2g, rh_w3, rh_b3, refb, Qcat);

  // query MLP
  k_gemm_bf16<8, 160, EPI_BIAS | EPI_GELU | EPI_WRB><<<gQ, blk, sh(8, 160), stream>>>(
      Qcat, qs_w1T, qs_b1, nullptr, Q1g, 128);
  k_gemm_bf16<8, 128, EPI_BIAS | EPI_WRB><<<gQ, blk, sh(8, 128), stream>>>(
      Q1g, qs_w2T, qs_b2, nullptr, Qry, 128);

  // value path
  k_gemm_bf16<8, 256, EPI_BIAS | EPI_WRB><<<gV, blk, sh(8, 256), stream>>>(
      Va, pv_wT, pv_b, nullptr, Vg, 128);
  k_gemm_bf16<8, 128, EPI_BIAS | EPI_WRF><<<gV, blk, sh(8, 128), stream>>>(
      Vg, vp_wT, vp_b, value, nullptr, 128);

  // sampling offsets + attention weights
  k_gemm_bf16<6, 128, EPI_BIAS | EPI_WRF><<<gQ, blk, sh(6, 128), stream>>>(
      Qry, so_wT, so_b, offsb, nullptr, 96);
  k_gemm_bf16<3, 128, EPI_BIAS | EPI_WRF><<<gQ, blk, sh(3, 128), stream>>>(
      Qry, aw_wT, aw_b, attl, nullptr, 48);
  k_softmax_p<<<(kRows * kHeads + 255) / 256, blk, 0, stream>>>(attl);

  // deformable bilinear sampling -> bf16
  k_msda<<<(kRows * kD + 255) / 256, blk, 0, stream>>>(value, refb, offsb, attl, outb);

  // output projections
  k_gemm_bf16<8, 128, EPI_BIAS | EPI_WRB><<<gQ, blk, sh(8, 128), stream>>>(
      outb, op_wT, op_b, nullptr, o2b, 128);
  k_gemm_bf16<8, 128, EPI_BIAS | EPI_WRF><<<gQ, blk, sh(8, 128), stream>>>(
      o2b, po_wT, po_b, (float*)d_out, nullptr, 128);
}
